// CustomStridedAttention_86062554677562
// MI455X (gfx1250) — matrix operand order repaired and checked by reading, not compiled, not run on hardware
//
#include <hip/hip_runtime.h>
#include <stdint.h>

typedef __bf16 bf16;
typedef __attribute__((ext_vector_type(16))) __bf16 v16bf;
typedef __attribute__((ext_vector_type(8)))  float  v8f;

union FragAB { v16bf v; uint4 q[2]; };

#define WMMA_BF16(a, b, c) \
  __builtin_amdgcn_wmma_f32_16x16x32_bf16(false, (a), false, (b), (short)0, (c), false, false)

#define B_   2
#define S_   4096
#define E_   1024
#define H_   16
#define ST_  8
#define DH_  64
#define NT_  512            // tokens per residue group (S_/ST_)
#define MROWS (B_*S_)       // 8192
#define NQKV  (3*E_)        // 3072

__device__ __forceinline__ v8f vzero8() {
  v8f z = {0.f,0.f,0.f,0.f,0.f,0.f,0.f,0.f};
  return z;
}

// ---- CDNA5 async global->LDS copy (ASYNCcnt-tracked, no VGPR staging)
__device__ __forceinline__ void async_b128(uint32_t lds_off, const void* g) {
  asm volatile("global_load_async_to_lds_b128 %0, %1, off"
               :: "v"(lds_off), "v"((unsigned long long)(uintptr_t)g)
               : "memory");
}
__device__ __forceinline__ void wait_async0() {
  asm volatile("s_wait_asynccnt 0x0" ::: "memory");
}
__device__ __forceinline__ void wait_async4() {
  asm volatile("s_wait_asynccnt 0x4" ::: "memory");
}

// ---------------------------------------------------------------- converts
__global__ void k_cvt_f32_bf16(const float* __restrict__ src,
                               bf16* __restrict__ dst, int n) {
  int i = (blockIdx.x * blockDim.x + threadIdx.x) * 4;
  if (i < n) {
    float4 f = *reinterpret_cast<const float4*>(src + i);
    union { bf16 h[4]; uint2 u2; } o;
    o.h[0] = (bf16)f.x; o.h[1] = (bf16)f.y; o.h[2] = (bf16)f.z; o.h[3] = (bf16)f.w;
    *reinterpret_cast<uint2*>(dst + i) = o.u2;
  }
}

// src [K][N] f32 -> dst [N][K] bf16 (tiled, coalesced both sides)
__global__ void k_transpose_cvt(const float* __restrict__ src,
                                bf16* __restrict__ dst, int K, int N) {
  __shared__ float tile[32][33];
  int nb = blockIdx.x * 32, kb = blockIdx.y * 32;
  int tx = threadIdx.x, ty = threadIdx.y;   // 32 x 8
#pragma unroll
  for (int j = 0; j < 32; j += 8)
    tile[ty + j][tx] = src[(size_t)(kb + ty + j) * N + nb + tx];
  __syncthreads();
#pragma unroll
  for (int j = 0; j < 32; j += 8)
    dst[(size_t)(nb + ty + j) * K + kb + tx] = (bf16)tile[tx][ty + j];
}

// ---------------------------------------------------------------- GEMM core
// 128x128 tile, 8 waves (2x4), each wave 64x32 = 4x2 WMMA subtiles, K-step 32.
// Double-buffered LDS fed by async global->LDS (ASYNCcnt).
__device__ __forceinline__ void gemm_mainloop(const bf16* __restrict__ X,
                                              const bf16* __restrict__ Wt,
                                              int tM, int tN, int K,
                                              v8f acc[4][2]) {
  constexpr int LDT = 40;                      // +8 bf16 pad: conflict-free
  __shared__ bf16 As[2][128 * LDT];
  __shared__ bf16 Bs[2][128 * LDT];
  const int tid = threadIdx.x;
  const int w = tid >> 5, lane = tid & 31;
  const int wm = w & 1, wn = w >> 1;
  const int lrow = lane & 15, lhalf = lane >> 4;
  const int ldRow = tid >> 1, ldHalf = (tid & 1) * 16;

  const bf16* gA = X + (size_t)(tM + ldRow) * K + ldHalf;
  const bf16* gB = Wt + (size_t)(tN + ldRow) * K + ldHalf;
  const uint32_t lA0 = (uint32_t)(uintptr_t)&As[0][ldRow * LDT + ldHalf];
  const uint32_t lB0 = (uint32_t)(uintptr_t)&Bs[0][ldRow * LDT + ldHalf];
  const uint32_t bufStride = (uint32_t)(128 * LDT * sizeof(bf16));

  auto issue = [&](int buf, int k0) {
    uint32_t la = lA0 + buf * bufStride;
    uint32_t lb = lB0 + buf * bufStride;
    async_b128(la,      gA + k0);
    async_b128(la + 16, gA + k0 + 8);
    async_b128(lb,      gB + k0);
    async_b128(lb + 16, gB + k0 + 8);
  };

  const int KT = K / 32;
  issue(0, 0);
  for (int kt = 0; kt < KT; kt++) {
    const int cur = kt & 1;
    if (kt + 1 < KT) { issue(cur ^ 1, (kt + 1) * 32); wait_async4(); }
    else             { wait_async0(); }
    __syncthreads();                       // tile `cur` visible to all waves
    const bf16* Ac = As[cur];
    const bf16* Bc = Bs[cur];
    FragAB a[4], b[2];
#pragma unroll
    for (int mi = 0; mi < 4; mi++) {
      int ar = wm * 64 + mi * 16 + lrow;
      a[mi].q[0] = *reinterpret_cast<const uint4*>(&Ac[ar * LDT + lhalf * 8]);
      a[mi].q[1] = *reinterpret_cast<const uint4*>(&Ac[ar * LDT + lhalf * 8 + 16]);
    }
#pragma unroll
    for (int ni = 0; ni < 2; ni++) {
      int bc = wn * 32 + ni * 16 + lrow;
      b[ni].q[0] = *reinterpret_cast<const uint4*>(&Bc[bc * LDT + lhalf * 8]);
      b[ni].q[1] = *reinterpret_cast<const uint4*>(&Bc[bc * LDT + lhalf * 8 + 16]);
    }
#pragma unroll
    for (int mi = 0; mi < 4; mi++)
#pragma unroll
      for (int ni = 0; ni < 2; ni++)
        acc[mi][ni] = WMMA_BF16(a[mi].v, b[ni].v, acc[mi][ni]);
    __syncthreads();                       // all reads done before overwrite
  }
}

// ---------------------------------------------------------------- GEMM1: QKV
__global__ __launch_bounds__(256) void k_gemm_qkv(
    const bf16* __restrict__ X, const bf16* __restrict__ Wt,
    const float* __restrict__ bias,
    bf16* __restrict__ Qr, bf16* __restrict__ Kr, bf16* __restrict__ Vt) {
  const int mt = blockIdx.x % (MROWS / 128);
  const int nt = blockIdx.x / (MROWS / 128);
  const int tM = mt * 128, tN = nt * 128;
  const int tid = threadIdx.x;
  const int w = tid >> 5, lane = tid & 31;
  const int wm = w & 1, wn = w >> 1;
  const int lrow = lane & 15, lhalf = lane >> 4;

  v8f acc[4][2];
#pragma unroll
  for (int i = 0; i < 4; i++)
#pragma unroll
    for (int j = 0; j < 2; j++) acc[i][j] = vzero8();

  gemm_mainloop(X, Wt, tM, tN, E_, acc);

  // scatter epilogue into attention-friendly layouts
#pragma unroll
  for (int mi = 0; mi < 4; mi++)
#pragma unroll
    for (int ni = 0; ni < 2; ni++)
#pragma unroll
      for (int r = 0; r < 8; r++) {
        int Mg = tM + wm * 64 + mi * 16 + lhalf * 8 + r;
        int Ng = tN + wn * 32 + ni * 16 + lrow;
        float val = acc[mi][ni][r] + bias[Ng];
        int bb = Mg >> 12, s = Mg & 4095;
        int g = s & 7, t = s >> 3;
        int sec = Ng >> 10, rem = Ng & 1023;
        int hh = rem >> 6, d = rem & 63;
        int grp = (bb * H_ + hh) * ST_ + g;
        bf16 bvv = (bf16)val;
        if (sec == 0)      Qr[((size_t)grp * NT_ + t) * DH_ + d] = bvv;
        else if (sec == 1) Kr[((size_t)grp * NT_ + t) * DH_ + d] = bvv;
        else               Vt[((size_t)grp * DH_ + d) * NT_ + t] = bvv;
      }
}

// ---------------------------------------------------------------- attention
__global__ __launch_bounds__(256) void k_attn(
    const bf16* __restrict__ Qr, const bf16* __restrict__ Kr,
    const bf16* __restrict__ Vt, bf16* __restrict__ A) {
  extern __shared__ char smem[];
  bf16* Ks  = reinterpret_cast<bf16*>(smem);   // [512][64]
  bf16* Vts = Ks + NT_ * DH_;                  // [64][512]
  bf16* Ps  = Vts + DH_ * NT_;                 // [8][16*32] per-wave bounce
  const int gid = blockIdx.x;
  const int b  = gid / (H_ * ST_);
  const int h  = (gid / ST_) % H_;
  const int g  = gid % ST_;
  const int tid = threadIdx.x, w = tid >> 5, lane = tid & 31;
  const int lrow = lane & 15, lhalf = lane >> 4;

  { // stage K and V^T into LDS via async copy engine
    const bf16* kg = Kr + (size_t)gid * NT_ * DH_;
    const bf16* vg = Vt + (size_t)gid * NT_ * DH_;
    uint32_t kl = (uint32_t)(uintptr_t)Ks;
    uint32_t vl = (uint32_t)(uintptr_t)Vts;
    for (int i = tid * 8; i < NT_ * DH_; i += 256 * 8) {
      async_b128(kl + i * 2, kg + i);
      async_b128(vl + i * 2, vg + i);
    }
    wait_async0();
  }
  __syncthreads();

  bf16* myP = Ps + w * (16 * 32);

  for (int qb = w; qb < NT_ / 16; qb += 8) {
    FragAB qa0, qa1;   // Q rows qb*16.., d 0-31 and 32-63
    const bf16* qbase = Qr + ((size_t)gid * NT_ + qb * 16 + lrow) * DH_;
    qa0.q[0] = *reinterpret_cast<const uint4*>(qbase + lhalf * 8);
    qa0.q[1] = *reinterpret_cast<const uint4*>(qbase + lhalf * 8 + 16);
    qa1.q[0] = *reinterpret_cast<const uint4*>(qbase + 32 + lhalf * 8);
    qa1.q[1] = *reinterpret_cast<const uint4*>(qbase + 32 + lhalf * 8 + 16);

    v8f oacc[4];
#pragma unroll
    for (int f = 0; f < 4; f++) oacc[f] = vzero8();
    float mrun[8], lsum[8];
#pragma unroll
    for (int r = 0; r < 8; r++) { mrun[r] = -3.0e38f; lsum[r] = 0.f; }

    for (int kb = 0; kb < NT_ / 32; kb++) {
      // ---- scores S[16 x 32] = Q @ K^T * scale
      v8f s0 = vzero8(), s1 = vzero8();
      {
        FragAB bl, bh;
        int t0 = (kb * 32 + lrow) * DH_;
        bl.q[0] = *reinterpret_cast<const uint4*>(&Ks[t0 + lhalf * 8]);
        bl.q[1] = *reinterpret_cast<const uint4*>(&Ks[t0 + lhalf * 8 + 16]);
        bh.q[0] = *reinterpret_cast<const uint4*>(&Ks[t0 + 32 + lhalf * 8]);
        bh.q[1] = *reinterpret_cast<const uint4*>(&Ks[t0 + 32 + lhalf * 8 + 16]);
        s0 = WMMA_BF16(qa0.v, bl.v, s0);
        s0 = WMMA_BF16(qa1.v, bh.v, s0);
        int t1 = (kb * 32 + 16 + lrow) * DH_;
        bl.q[0] = *reinterpret_cast<const uint4*>(&Ks[t1 + lhalf * 8]);
        bl.q[1] = *reinterpret_cast<const uint4*>(&Ks[t1 + lhalf * 8 + 16]);
        bh.q[0] = *reinterpret_cast<const uint4*>(&Ks[t1 + 32 + lhalf * 8]);
        bh.q[1] = *reinterpret_cast<const uint4*>(&Ks[t1 + 32 + lhalf * 8 + 16]);
        s1 = WMMA_BF16(qa0.v, bl.v, s1);
        s1 = WMMA_BF16(qa1.v, bh.v, s1);
      }
      // ---- online softmax over this 32-column block
      float p0e[8], p1e[8];
#pragma unroll
      for (int r = 0; r < 8; r++) { p0e[r] = s0[r] * 0.125f; p1e[r] = s1[r] * 0.125f; }
#pragma unroll
      for (int r = 0; r < 8; r++) {
        float mx = fmaxf(p0e[r], p1e[r]);
#pragma unroll
        for (int msk = 8; msk >= 1; msk >>= 1) mx = fmaxf(mx, __shfl_xor(mx, msk, 16));
        float mnew = fmaxf(mrun[r], mx);
        float corr = __expf(mrun[r] - mnew);
        mrun[r] = mnew;
        lsum[r] *= corr;
#pragma unroll
        for (int f = 0; f < 4; f++) oacc[f][r] *= corr;
        p0e[r] = __expf(p0e[r] - mnew);
        p1e[r] = __expf(p1e[r] - mnew);
        float rs = p0e[r] + p1e[r];
#pragma unroll
        for (int msk = 8; msk >= 1; msk >>= 1) rs += __shfl_xor(rs, msk, 16);
        lsum[r] += rs;
      }
      // ---- bounce P through wave-private LDS: C-layout -> A-layout
#pragma unroll
      for (int r = 0; r < 8; r++) {
        int Mr = lhalf * 8 + r;
        myP[Mr * 32 + lrow]      = (bf16)p0e[r];
        myP[Mr * 32 + 16 + lrow] = (bf16)p1e[r];
      }
      asm volatile("s_wait_dscnt 0" ::: "memory");
      FragAB pa;
      pa.q[0] = *reinterpret_cast<const uint4*>(&myP[lrow * 32 + lhalf * 8]);
      pa.q[1] = *reinterpret_cast<const uint4*>(&myP[lrow * 32 + lhalf * 8 + 16]);
      asm volatile("" ::: "memory");
      // ---- O += P @ V_block
#pragma unroll
      for (int f = 0; f < 4; f++) {
        FragAB vb;
        int d = f * 16 + lrow;
        vb.q[0] = *reinterpret_cast<const uint4*>(&Vts[d * NT_ + kb * 32 + lhalf * 8]);
        vb.q[1] = *reinterpret_cast<const uint4*>(&Vts[d * NT_ + kb * 32 + lhalf * 8 + 16]);
        oacc[f] = WMMA_BF16(pa.v, vb.v, oacc[f]);
      }
    }
    // ---- normalize + inverse regroup store (s = g + 8*t)
#pragma unroll
    for (int r = 0; r < 8; r++) {
      float inv = 1.0f / lsum[r];
      int t = qb * 16 + lhalf * 8 + r;
      int s = g + ST_ * t;
      size_t rowoff = ((size_t)b * S_ + s) * E_ + h * DH_;
#pragma unroll
      for (int f = 0; f < 4; f++)
        A[rowoff + f * 16 + lrow] = (bf16)(oacc[f][r] * inv);
    }
  }
}

// ---------------------------------------------------------------- GEMM2: out
__global__ __launch_bounds__(256) void k_gemm_out(
    const bf16* __restrict__ X, const bf16* __restrict__ Wt,
    const float* __restrict__ bias, float* __restrict__ out) {
  const int mt = blockIdx.x % (MROWS / 128);
  const int nt = blockIdx.x / (MROWS / 128);
  const int tM = mt * 128, tN = nt * 128;
  const int tid = threadIdx.x;
  const int w = tid >> 5, lane = tid & 31;
  const int wm = w & 1, wn = w >> 1;
  const int lrow = lane & 15, lhalf = lane >> 4;

  v8f acc[4][2];
#pragma unroll
  for (int i = 0; i < 4; i++)
#pragma unroll
    for (int j = 0; j < 2; j++) acc[i][j] = vzero8();

  gemm_mainloop(X, Wt, tM, tN, E_, acc);

#pragma unroll
  for (int mi = 0; mi < 4; mi++)
#pragma unroll
    for (int ni = 0; ni < 2; ni++)
#pragma unroll
      for (int r = 0; r < 8; r++) {
        int Mg = tM + wm * 64 + mi * 16 + lhalf * 8 + r;
        int Ng = tN + wn * 32 + ni * 16 + lrow;
        out[(size_t)Mg * E_ + Ng] = acc[mi][ni][r] + bias[Ng];
      }
}

// ---------------------------------------------------------------- launcher
extern "C" void kernel_launch(void* const* d_in, const int* in_sizes, int n_in,
                              void* d_out, int out_size, void* d_ws, size_t ws_size,
                              hipStream_t stream) {
  const float* x    = (const float*)d_in[0];
  const float* Wqkv = (const float*)d_in[1];
  const float* bqkv = (const float*)d_in[2];
  const float* Wout = (const float*)d_in[3];
  const float* bout = (const float*)d_in[4];
  float* out = (float*)d_out;

  char* ws = (char*)d_ws;
  bf16* Xbf   = (bf16*)(ws);                          // 16 MB
  bf16* Wqkvt = (bf16*)(ws + ((size_t)16 << 20));     //  6 MB
  bf16* Woutt = (bf16*)(ws + ((size_t)22 << 20));     //  2 MB
  bf16* Qr    = (bf16*)(ws + ((size_t)24 << 20));     // 16 MB
  bf16* Kr    = (bf16*)(ws + ((size_t)40 << 20));     // 16 MB
  bf16* Vtg   = (bf16*)(ws + ((size_t)56 << 20));     // 16 MB
  bf16* Abf   = (bf16*)(ws + ((size_t)72 << 20));     // 16 MB

  int nx = MROWS * E_;
  k_cvt_f32_bf16<<<(nx / 4 + 255) / 256, 256, 0, stream>>>(x, Xbf, nx);
  k_transpose_cvt<<<dim3(NQKV / 32, E_ / 32), dim3(32, 8), 0, stream>>>(Wqkv, Wqkvt, E_, NQKV);
  k_transpose_cvt<<<dim3(E_ / 32, E_ / 32), dim3(32, 8), 0, stream>>>(Wout, Woutt, E_, E_);
  k_gemm_qkv<<<(MROWS / 128) * (NQKV / 128), 256, 0, stream>>>(Xbf, Wqkvt, bqkv, Qr, Kr, Vtg);
  k_attn<<<B_ * H_ * ST_, 256, 139264, stream>>>(Qr, Kr, Vtg, Abf);
  k_gemm_out<<<(MROWS / 128) * (E_ / 128), 256, 0, stream>>>(Abf, Woutt, bout, out);
}
